// NeuralSDEGenerator_31774168056080
// MI455X (gfx1250) — matrix operand order repaired and checked by reading, not compiled, not run on hardware
//
#include <hip/hip_runtime.h>
#include <hip/hip_bf16.h>
#include <math.h>

// ---------------------------------------------------------------------------
// NeuralSDEGenerator for MI455X (gfx1250, wave32, WMMA + async-to-LDS).
// B=1024, L=128, D_IN=5, H=64, RES=256, BD=8, D_OUT=1
// ---------------------------------------------------------------------------

typedef __attribute__((ext_vector_type(16))) __bf16          v16bf;
typedef __attribute__((ext_vector_type(8)))  float           v8f;
typedef __attribute__((ext_vector_type(8)))  unsigned short  v8us;

union Frag16 { v16bf v; v8us u[2]; };

#define BSZ   1024
#define LSZ   128
#define HSZ   64
#define RESZ  256
#define BDIM  8
#define NGRP  9            // 1 drift (B1) + 8 diffusion (B2[k])
#define NCOL  (NGRP*RESZ)  // 2304 fused output columns
#define NT    (NCOL/16)    // 144 N-tiles
#define KB    (RESZ/32)    // 8 K-blocks of 32
#define NFRAG (NT*KB)      // 1152 B-fragments, 1024 B each (bf16)
#define NWAVE 16           // 512 threads/block

// per-wave LDS staging: 2 buffers x 9 fragments x 1024 B
#define WSTG_BUF   9216                   // bytes per buffer
#define WSTG_WAVE  (2*WSTG_BUF)           // 18432 bytes per wave
#define WSTG_TOTAL (NWAVE*WSTG_WAVE)      // 294912 bytes (dynamic LDS)

// workspace layout (bytes)
#define WS_WFRAG 0
#define WS_BIAS  (NFRAG*1024)           // 1,179,648
#define WS_V     (WS_BIAS + NCOL*4)     // + 9,216

__device__ __forceinline__ unsigned short f2bf(float f) {
    unsigned int u = __float_as_uint(f);
    u += 0x7FFFu + ((u >> 16) & 1u);     // round-to-nearest-even
    return (unsigned short)(u >> 16);
}

// sum across the 16 lanes of a half-wave (butterfly)
__device__ __forceinline__ float hredux16(float v) {
#pragma unroll
    for (int m = 8; m >= 1; m >>= 1) v += __shfl_xor(v, m, 16);
    return v;
}

// Async L2->LDS copy of one fragment (2 x 16 B/lane = 1 KB/wave).
// global addr = sbase + voff (+16); LDS addr = lds_addr (+16).
__device__ __forceinline__ void async_ld_frag(unsigned lds_addr, unsigned voff,
                                              unsigned long long sbase) {
    asm volatile("global_load_async_to_lds_b128 %0, %1, %2"
                 :: "v"(lds_addr), "v"(voff), "s"(sbase));
    asm volatile("global_load_async_to_lds_b128 %0, %1, %2 offset:16"
                 :: "v"(lds_addr), "v"(voff), "s"(sbase));
}

// ---------------------------------------------------------------------------
// Prep 1: fold rho1/rho3 into B1/B2 and swizzle to bf16 WMMA B-fragment layout.
// Fragment f = ntile*8 + kb holds the 32(K)x16(N) tile; lane L stores its 16
// elements contiguously at f*1024 + L*32 bytes, element e <-> K = (L/16)*16+e.
// ---------------------------------------------------------------------------
__global__ __launch_bounds__(256) void prep_weights(
    const float* __restrict__ B1, const float* __restrict__ B2,
    const float* __restrict__ rho1p, const float* __restrict__ rho3p,
    unsigned short* __restrict__ Wfrag)
{
    int gid  = blockIdx.x * 256 + threadIdx.x;   // 0..36863  (1152 frags x 32 lanes)
    int frag = gid >> 5;
    int L    = gid & 31;
    int ntile = frag >> 3;
    int kb    = frag & 7;
    int c = ntile * 16 + (L & 15);               // fused output column 0..2303
    int g = c >> 8;                              // group: 0=drift, 1..8=B2[k]
    int i = c & 255;
    float s = (g == 0) ? rho1p[0] : rho3p[0];
    const float* wrow = (g == 0) ? (B1 + (size_t)i * RESZ)
                                 : (B2 + ((size_t)(g - 1) * RESZ + i) * RESZ);
    int kbase = kb * 32 + (L >> 4) * 16;
    unsigned short* dst = Wfrag + (size_t)frag * 512 + L * 16;
#pragma unroll
    for (int e = 0; e < 16; ++e)
        dst[e] = f2bf(s * wrow[kbase + e]);
}

// ---------------------------------------------------------------------------
// Prep 2: fused bias  bias[c] = g==0 ? rho2*l1[i] : rho4*l2[k][i]
// ---------------------------------------------------------------------------
__global__ void prep_bias(const float* __restrict__ l1, const float* __restrict__ l2,
                          const float* __restrict__ rho2p, const float* __restrict__ rho4p,
                          float* __restrict__ bias)
{
    int c = blockIdx.x * 256 + threadIdx.x;
    if (c >= NCOL) return;
    int g = c >> 8, i = c & 255;
    bias[c] = (g == 0) ? rho2p[0] * l1[i] : rho4p[0] * l2[(g - 1) * RESZ + i];
}

// ---------------------------------------------------------------------------
// Prep 3: initial state  V = tanh(V_noise @ W1^T + b1) @ W2^T + b2
// ---------------------------------------------------------------------------
__global__ __launch_bounds__(64) void prep_v(
    const float* __restrict__ Vn, const float* __restrict__ W1,
    const float* __restrict__ b1, const float* __restrict__ W2,
    const float* __restrict__ b2, float* __restrict__ V)
{
    __shared__ float hsh[HSZ];
    int b = blockIdx.x, j = threadIdx.x;
    float a = b1[j];
#pragma unroll
    for (int d = 0; d < 5; ++d) a += Vn[b * 5 + d] * W1[j * 5 + d];
    hsh[j] = tanhf(a);
    __syncthreads();
    for (int i = j; i < RESZ; i += HSZ) {
        float s = b2[i];
#pragma unroll 8
        for (int q = 0; q < HSZ; ++q) s += hsh[q] * W2[i * HSZ + q];
        V[(size_t)b * RESZ + i] = s;
    }
}

// ---------------------------------------------------------------------------
// Main scan kernel: 64 blocks x 512 threads (16 waves of 32, 4 waves/SIMD).
// Block owns 16 batch rows for all 127 steps; wave w owns state columns
// [w*16, w*16+16) => 9 C/D tiles (9 groups) in registers (72 acc VGPRs).
// B fragments stream L2 -> LDS one K-block ahead via async copies (ASYNCcnt);
// WMMAs consume them from LDS with ds_load_b128 pipelined two tiles ahead.
// ---------------------------------------------------------------------------
__global__ __launch_bounds__(512, 1) void sde_scan(
    const float* __restrict__ inc,  const float* __restrict__ Wr,
    const float* __restrict__ brp,  const float* __restrict__ rho5p,
    const float* __restrict__ V,    const unsigned short* __restrict__ Wfrag,
    const float* __restrict__ bias, float* __restrict__ out)
{
    __shared__ unsigned short Abuf[16 * RESZ];  // state as bf16, [m][K], 8 KB
    __shared__ float sdw[BDIM * 16];            // rho5*dW, [k][m]
    __shared__ float oacc[16];                  // Wr readout accumulator
    extern __shared__ unsigned char Bstage[];   // 288 KB async B staging

    const int tid = threadIdx.x;
    const int w   = tid >> 5;       // wave 0..15
    const int L   = tid & 31;       // lane
    const int h   = L >> 4;         // lane half
    const int n   = L & 15;         // N within tile / M for A row
    const int b0  = blockIdx.x * 16;

    const float rho5 = rho5p[0];
    const float brv  = brp[0];

    const int   col = w * 16 + n;   // this lane's state column
    const float wrv = Wr[col];
    float biasr[NGRP];
#pragma unroll
    for (int g = 0; g < NGRP; ++g) biasr[g] = bias[g * RESZ + col];

    // Load initial state in C/D fragment layout: element j <-> row m = j + 8*h.
    float r[8];
#pragma unroll
    for (int j = 0; j < 8; ++j)
        r[j] = V[(size_t)(b0 + j + 8 * h) * RESZ + col];

    // t = 0 readout (cross-lane reduce over the 16 columns per half-wave)
    if (tid < 16) oacc[tid] = 0.f;
    __syncthreads();
    {
        float p[8];
#pragma unroll
        for (int j = 0; j < 8; ++j) p[j] = hredux16(r[j] * wrv);
        if (n == 0)
#pragma unroll
            for (int j = 0; j < 8; ++j) atomicAdd(&oacc[j + 8 * h], p[j]);
    }
    __syncthreads();
    if (tid < 16) out[(size_t)(b0 + tid) * LSZ + 0] = oacc[tid] + brv;

    // Async addressing pieces:
    //   global frag byte = Wfrag + g*131072 + w*8192 + kb*1024 + L*32
    //   -> SGPR base carries (Wfrag, g, kb); VGPR voff carries (w, L).
    unsigned long long wfu = (unsigned long long)(uintptr_t)Wfrag;
    const unsigned voff = (unsigned)(w * 8192 + L * 32);
    const unsigned ldsWbase =
        (unsigned)(uintptr_t)Bstage + (unsigned)(w * WSTG_WAVE + L * 32);

    for (int t = 0; t < LSZ - 1; ++t) {
        asm volatile("" : "+s"(wfu));   // keep base scalar + defeat LICM

        // Kick off K-block 0 into buffer 0 (overlaps with A/sdw staging).
#pragma unroll
        for (int g = 0; g < NGRP; ++g)
            async_ld_frag(ldsWbase + (unsigned)(g * 1024), voff,
                          wfu + (unsigned long long)(g * 131072));

        __syncthreads();
        // Stage state as bf16 into shared A-buffer (each wave: its 16 columns).
#pragma unroll
        for (int j = 0; j < 8; ++j)
            Abuf[(j + 8 * h) * RESZ + col] = f2bf(r[j]);
        // Stage rho5*dW tile:  dW[t]=inc[0]+inc[1] at t=0, else inc[t+1].
        if (tid < 128) {
            int k = tid & 7, m = tid >> 3;
            const float* ip = inc + (size_t)(b0 + m) * (LSZ * BDIM);
            float dv = (t == 0) ? (ip[k] + ip[BDIM + k]) : ip[(t + 1) * BDIM + k];
            sdw[k * 16 + m] = rho5 * dv;
        }
        if (tid < 16) oacc[tid] = 0.f;
        __syncthreads();

        // ---- 16x2304 = r(16x256) @ Wbig(256x2304); 72 WMMAs per wave ----
        v8f acc[NGRP];
#pragma unroll
        for (int g = 0; g < NGRP; ++g) acc[g] = (v8f)0.f;

#pragma unroll
        for (int kb = 0; kb < KB; ++kb) {
            const int buf = kb & 1;
            // Current K-block was issued one full compute-block ago: drain it.
            // (in-order completion; counter 0 after this, so the 18 copies
            //  issued below never hit the ASYNCcnt=63 issue-stall cap)
            asm volatile("s_wait_asynccnt 0" ::: "memory");

            // A fragment from LDS: row m = n, K = kb*32 + h*16 + 0..15
            Frag16 a;
            const v8us* ap = (const v8us*)&Abuf[n * RESZ + kb * 32 + h * 16];
            a.u[0] = ap[0];   // K = kb*32 + h*16 + 0..7
            a.u[1] = ap[1];   // K = kb*32 + h*16 + 8..15

            // Issue next K-block copies into the other buffer; they overlap
            // with the 9 WMMAs below.
            if (kb < KB - 1) {
#pragma unroll
                for (int g = 0; g < NGRP; ++g)
                    async_ld_frag(
                        ldsWbase + (unsigned)((buf ^ 1) * WSTG_BUF + g * 1024),
                        voff,
                        wfu + (unsigned long long)(g * 131072 + (kb + 1) * 1024));
            }

            // B ds-reads pipelined two tiles ahead of the consuming WMMA.
            const unsigned char* bbase =
                Bstage + (size_t)w * WSTG_WAVE + buf * WSTG_BUF + L * 32;
            Frag16 f0, f1;
            {
                const v8us* bp0 = (const v8us*)bbase;
                f0.u[0] = bp0[0]; f0.u[1] = bp0[1];
                const v8us* bp1 = (const v8us*)(bbase + 1024);
                f1.u[0] = bp1[0]; f1.u[1] = bp1[1];
            }
#pragma unroll
            for (int g = 0; g < NGRP; ++g) {
                Frag16 f2;
                if (g < NGRP - 2) {
                    const v8us* bp = (const v8us*)(bbase + (g + 2) * 1024);
                    f2.u[0] = bp[0]; f2.u[1] = bp[1];
                } else {
                    f2 = f0;
                }
                acc[g] = __builtin_amdgcn_wmma_f32_16x16x32_bf16(
                    false, a.v, false, f0.v, (short)0, acc[g], false, false);
                f0 = f1;
                f1 = f2;
            }
        }

        // ---- epilogue (in place): r += tanh(drift) + sum_k sdw*tanh(diff) ----
#pragma unroll
        for (int j = 0; j < 8; ++j)
            r[j] += tanhf(acc[0][j] + biasr[0]);
#pragma unroll
        for (int k = 0; k < BDIM; ++k) {
            float sd[8];
#pragma unroll
            for (int j = 0; j < 8; ++j) sd[j] = sdw[k * 16 + 8 * h + j];
#pragma unroll
            for (int j = 0; j < 8; ++j)
                r[j] += sd[j] * tanhf(acc[k + 1][j] + biasr[k + 1]);
        }

        // readout for step t+1 (cross-lane reduce, then 8 atomics per half-wave)
        {
            float p[8];
#pragma unroll
            for (int j = 0; j < 8; ++j) p[j] = hredux16(r[j] * wrv);
            if (n == 0)
#pragma unroll
                for (int j = 0; j < 8; ++j) atomicAdd(&oacc[j + 8 * h], p[j]);
        }
        __syncthreads();
        if (tid < 16) out[(size_t)(b0 + tid) * LSZ + (t + 1)] = oacc[tid] + brv;
    }
}

// ---------------------------------------------------------------------------
extern "C" void kernel_launch(void* const* d_in, const int* in_sizes, int n_in,
                              void* d_out, int out_size, void* d_ws, size_t ws_size,
                              hipStream_t stream)
{
    (void)in_sizes; (void)n_in; (void)out_size; (void)ws_size;
    const float* Vn  = (const float*)d_in[0];
    const float* inc = (const float*)d_in[1];
    const float* W1  = (const float*)d_in[2];
    const float* b1  = (const float*)d_in[3];
    const float* W2  = (const float*)d_in[4];
    const float* b2  = (const float*)d_in[5];
    const float* r1  = (const float*)d_in[6];
    const float* r2  = (const float*)d_in[7];
    const float* r3  = (const float*)d_in[8];
    const float* r4  = (const float*)d_in[9];
    const float* r5  = (const float*)d_in[10];
    const float* B1  = (const float*)d_in[11];
    const float* B2  = (const float*)d_in[12];
    const float* l1  = (const float*)d_in[13];
    const float* l2  = (const float*)d_in[14];
    const float* Wr  = (const float*)d_in[15];
    const float* br  = (const float*)d_in[16];
    float* out = (float*)d_out;

    char* ws = (char*)d_ws;
    unsigned short* Wfrag = (unsigned short*)(ws + WS_WFRAG);
    float*          bias  = (float*)(ws + WS_BIAS);
    float*          V     = (float*)(ws + WS_V);

    prep_weights<<<NFRAG * 32 / 256, 256, 0, stream>>>(B1, B2, r1, r3, Wfrag);
    prep_bias<<<(NCOL + 255) / 256, 256, 0, stream>>>(l1, l2, r2, r4, bias);
    prep_v<<<BSZ, HSZ, 0, stream>>>(Vn, W1, b1, W2, b2, V);
    sde_scan<<<BSZ / 16, 512, WSTG_TOTAL, stream>>>(inc, Wr, br, r5, V, Wfrag, bias, out);
}
